// SplitNN_7095285973801
// MI455X (gfx1250) — matrix operand order repaired and checked by reading, not compiled, not run on hardware
//
#include <hip/hip_runtime.h>

typedef __attribute__((ext_vector_type(16))) _Float16 v16h;
typedef __attribute__((ext_vector_type(8)))  _Float16 v8h;
typedef __attribute__((ext_vector_type(8)))  float    v8f;
typedef __attribute__((ext_vector_type(4)))  float    v4f;

#define LCA_LAM    0.5f
#define LCA_INVTAU (1.0f/1000.0f)
#define LCA_ITERS  500

// ---------------------------------------------------------------------------
// Pack lateral kernel G[g,o,dy,dx] = sum_{c,ky,kx} w[g,c,dy+ky-4,dx+kx-4]*w[o,c,ky,kx]
// directly into the WMMA A-fragment layout:
//   gpack[((mt*STEPS+s)*32+lane)*16+e]  holds  A[M=mt*16+(lane&15)][K]  with
//   K = e + 16*(lane>>4),  k = s*32+K,  tap = k/OCH, i = k%OCH.
// K ordering is channel-innermost: k = tap*OCH + i  (tap = dy*9+dx).
// ---------------------------------------------------------------------------
template<int OCH, int CIN>
__global__ void pack_g_kernel(const float* __restrict__ w, _Float16* __restrict__ gpack)
{
    constexpr int MT    = OCH / 16;
    constexpr int TAPS  = (OCH == 16) ? 82 : 81;   // pad to even tap count for OCH=16
    constexpr int STEPS = TAPS * OCH / 32;
    constexpr int TOT   = MT * STEPS * 512;
    int idx = blockIdx.x * blockDim.x + threadIdx.x;
    if (idx >= TOT) return;
    int e    = idx & 15;
    int lane = (idx >> 4) & 31;
    int ss   = (idx >> 9) % STEPS;
    int mt   = (idx >> 9) / STEPS;
    int g2   = lane >> 4;
    int K    = e + 16 * g2;
    int k    = ss * 32 + K;
    int tap  = k / OCH;              // OCH is power of two -> shift
    int i    = k & (OCH - 1);
    int M    = mt * 16 + (lane & 15);
    float val = 0.f;
    if (tap < 81) {
        int dy = tap / 9, dx = tap - dy * 9;
        for (int c = 0; c < CIN; ++c)
            for (int ky = 0; ky < 5; ++ky) {
                int ay = dy + ky - 4;
                if (ay < 0 || ay >= 5) continue;
                for (int kx = 0; kx < 5; ++kx) {
                    int ax = dx + kx - 4;
                    if (ax < 0 || ax >= 5) continue;
                    val += w[((M * CIN + c) * 5 + ay) * 5 + ax] *
                           w[((i * CIN + c) * 5 + ky) * 5 + kx];
                }
            }
    }
    gpack[idx] = (_Float16)val;
}

// ---------------------------------------------------------------------------
// drive = conv2d(x, w, pad=2), written in permuted [b][p][o] layout so the LCA
// kernel's membrane update uses 128-bit vector loads.
// ---------------------------------------------------------------------------
template<int CIN, int COUT>
__global__ void conv_drive_kernel(const float* __restrict__ in, const float* __restrict__ w,
                                  float* __restrict__ drive_perm)
{
    constexpr int TOT = 8 * 1024 * COUT;
    int idx = blockIdx.x * blockDim.x + threadIdx.x;
    if (idx >= TOT) return;
    int o = idx % COUT;
    int p = (idx / COUT) & 1023;
    int b = idx / (COUT * 1024);
    int y = p >> 5, x = p & 31;
    float s = 0.f;
    for (int c = 0; c < CIN; ++c)
        for (int ky = 0; ky < 5; ++ky) {
            int iy = y + ky - 2;
            if (iy < 0 || iy >= 32) continue;
            for (int kx = 0; kx < 5; ++kx) {
                int ix = x + kx - 2;
                if (ix < 0 || ix >= 32) continue;
                s += in[((b * CIN + c) * 32 + iy) * 32 + ix] *
                     w[((o * CIN + c) * 5 + ky) * 5 + kx];
            }
        }
    drive_perm[idx] = s;   // [b][p][o]
}

// ---------------------------------------------------------------------------
// LCA solver: one workgroup per batch image, 8 waves, whole image + packed G
// resident in LDS, 500 iterations with only workgroup barriers.
//   Phase A: a = relu(u - lam) -> f16 a_pad[yy][xx][c] (halo stays zero)
//   Phase B: inhib = G (x) a via v_wmma_f32_16x16x32_f16 implicit GEMM
//   Phase C: u += (drive - u - inhib + a) / tau    (u, drive in [b][p][c])
// ---------------------------------------------------------------------------
template<int OCH>
__global__ __launch_bounds__(256, 1)
void lca_kernel(const float* __restrict__ drive,
                const _Float16* __restrict__ gpack,
                float* __restrict__ u,
                float* __restrict__ a_out)
{
    constexpr int MT    = OCH / 16;
    constexpr int TAPS  = (OCH == 16) ? 82 : 81;
    constexpr int STEPS = TAPS * OCH / 32;
    constexpr int ROWS  = 41, COLS = 48;           // padded image, halo >= 4 + ghost row
    constexpr int APAD  = ROWS * COLS * OCH;       // halves
    constexpr int GH    = MT * STEPS * 512;        // halves

    __shared__ __align__(32) _Float16 a_pad[APAD];
    __shared__ __align__(32) _Float16 gp[GH];

    const int b    = blockIdx.x;
    const int tid  = threadIdx.x;
    const int wave = tid >> 5, lane = tid & 31, col = lane & 15, hi = lane >> 4;

    // G -> LDS, zero a_pad (halo included; halo never rewritten)
    {
        const v4f* src = (const v4f*)gpack;
        v4f* dst = (v4f*)gp;
        for (int t = tid; t < GH / 8; t += 256) dst[t] = src[t];
    }
    {
        v4f* dst = (v4f*)a_pad;
        v4f z; z[0] = 0.f; z[1] = 0.f; z[2] = 0.f; z[3] = 0.f;
        for (int t = tid; t < APAD / 8; t += 256) dst[t] = z;
    }

    float* ub = u + b * 1024 * OCH;
    const float* db = drive + b * 1024 * OCH;

    int pbase[8], sbase[8], gb2[8];
    const int i0 = (OCH == 32) ? 16 * hi : 0;      // B-fragment channel base
    #pragma unroll
    for (int j = 0; j < 8; ++j) {
        int nt = wave * 8 + j;                     // this wave's 8 pixel tiles
        int y  = nt >> 1;
        int x0 = ((nt & 1) << 4) + col;
        pbase[j] = (nt * 16 + col) * OCH;                  // u/drive [b][p][c]
        sbase[j] = ((y + 4) * COLS + (x0 + 4)) * OCH;      // a_pad store base
        gb2[j]   = (y * COLS + x0) * OCH + i0;             // gather base (tap 0)
    }

    __syncthreads();

    #pragma unroll 1
    for (int it = 0; it < LCA_ITERS; ++it) {
        // ---------------- Phase A ----------------
        #pragma unroll
        for (int j = 0; j < 8; ++j) {
            #pragma unroll
            for (int mt = 0; mt < MT; ++mt) {
                const float* up = ub + pbase[j] + mt * 16 + 8 * hi;
                v4f u0 = *(const v4f*)up;
                v4f u1 = *(const v4f*)(up + 4);
                v8h a8;
                #pragma unroll
                for (int r = 0; r < 4; ++r) {
                    a8[r]     = (_Float16)fmaxf(u0[r] - LCA_LAM, 0.f);
                    a8[r + 4] = (_Float16)fmaxf(u1[r] - LCA_LAM, 0.f);
                }
                *(v8h*)&a_pad[sbase[j] + mt * 16 + 8 * hi] = a8;   // ds_store_b128
            }
        }
        __syncthreads();

        // ---------------- Phase B: implicit GEMM ----------------
        v8f acc[8][MT];
        #pragma unroll
        for (int j = 0; j < 8; ++j)
            #pragma unroll
            for (int mt = 0; mt < MT; ++mt)
                #pragma unroll
                for (int r = 0; r < 8; ++r) acc[j][mt][r] = 0.f;

        #pragma unroll 1
        for (int s = 0; s < STEPS; ++s) {
            v16h afr[MT];
            #pragma unroll
            for (int mt = 0; mt < MT; ++mt)
                afr[mt] = *(const v16h*)&gp[((mt * STEPS + s) * 32 + lane) * 16];

            int toff;
            if (OCH == 32) {               // one tap per 32-wide K step
                int dy = s / 9, dx = s - dy * 9;
                toff = (dy * COLS + dx) * OCH;
            } else {                       // two taps per step, one per lane group
                int t0 = 2 * s + hi;
                int dy = t0 / 9, dx = t0 - dy * 9;
                toff = (dy * COLS + dx) * OCH;
            }

            #pragma unroll
            for (int j = 0; j < 8; ++j) {
                v16h bfr = *(const v16h*)&a_pad[gb2[j] + toff];    // 2x ds_load_b128
                #pragma unroll
                for (int mt = 0; mt < MT; ++mt)
                    acc[j][mt] = __builtin_amdgcn_wmma_f32_16x16x32_f16(
                        false, afr[mt], false, bfr, (short)0, acc[j][mt], false, false);
            }
        }

        // ---------------- Phase C: membrane update ----------------
        #pragma unroll
        for (int j = 0; j < 8; ++j) {
            #pragma unroll
            for (int mt = 0; mt < MT; ++mt) {
                float* up = ub + pbase[j] + mt * 16 + 8 * hi;
                const float* dp = db + pbase[j] + mt * 16 + 8 * hi;
                v4f u0 = *(const v4f*)up;
                v4f u1 = *(const v4f*)(up + 4);
                v4f d0 = *(const v4f*)dp;
                v4f d1 = *(const v4f*)(dp + 4);
                v4f n0, n1;
                #pragma unroll
                for (int r = 0; r < 4; ++r) {
                    float a0 = fmaxf(u0[r] - LCA_LAM, 0.f);
                    float a1 = fmaxf(u1[r] - LCA_LAM, 0.f);
                    n0[r] = u0[r] + (d0[r] - u0[r] - acc[j][mt][r]     + a0) * LCA_INVTAU;
                    n1[r] = u1[r] + (d1[r] - u1[r] - acc[j][mt][r + 4] + a1) * LCA_INVTAU;
                }
                *(v4f*)up       = n0;
                *(v4f*)(up + 4) = n1;
            }
        }
        __syncthreads();
    }

    // final sparse code a = relu(u - lam), NCHW for downstream layers
    #pragma unroll
    for (int j = 0; j < 8; ++j) {
        const float* up0 = ub + pbase[j];
        int p = (wave * 8 + j) * 16 + col;
        #pragma unroll
        for (int mt = 0; mt < MT; ++mt) {
            #pragma unroll
            for (int r = 0; r < 8; ++r) {
                int c = mt * 16 + 8 * hi + r;
                a_out[(b * OCH + c) * 1024 + p] = fmaxf(up0[mt * 16 + 8 * hi + r] - LCA_LAM, 0.f);
            }
        }
    }
}

// ---------------------------------------------------------------------------
// Small layers
// ---------------------------------------------------------------------------
__global__ void bn_kernel(const float* __restrict__ in, const float* g, const float* be,
                          const float* m, const float* v, float* __restrict__ out,
                          int C, int HW, int total)
{
    int idx = blockIdx.x * blockDim.x + threadIdx.x;
    if (idx >= total) return;
    int c = (idx / HW) % C;
    float sc = g[c] * rsqrtf(v[c] + 1e-5f);
    out[idx] = in[idx] * sc + (be[c] - m[c] * sc);
}

__global__ void pool_bn_kernel(const float* __restrict__ in, const float* g, const float* be,
                               const float* m, const float* v, float* __restrict__ out,
                               int C, int H, int W, int total)
{
    int idx = blockIdx.x * blockDim.x + threadIdx.x;
    if (idx >= total) return;
    int OW = W >> 1, OH = H >> 1;
    int ox = idx % OW;
    int oy = (idx / OW) % OH;
    int c  = (idx / (OW * OH)) % C;
    int b  = idx / (OW * OH * C);
    const float* ip = in + ((b * C + c) * H + oy * 2) * W + ox * 2;
    float mx = fmaxf(fmaxf(ip[0], ip[1]), fmaxf(ip[W], ip[W + 1]));
    float sc = g[c] * rsqrtf(v[c] + 1e-5f);
    out[idx] = mx * sc + (be[c] - m[c] * sc);
}

__global__ void conv_bias_relu_kernel(const float* __restrict__ in, const float* __restrict__ w,
                                      const float* __restrict__ bias, float* __restrict__ out,
                                      int CIN, int COUT, int H, int W, int total)
{
    int idx = blockIdx.x * blockDim.x + threadIdx.x;
    if (idx >= total) return;
    int x = idx % W;
    int y = (idx / W) % H;
    int o = (idx / (W * H)) % COUT;
    int b = idx / (W * H * COUT);
    float s = bias[o];
    for (int c = 0; c < CIN; ++c)
        for (int ky = 0; ky < 5; ++ky) {
            int iy = y + ky - 2;
            if (iy < 0 || iy >= H) continue;
            for (int kx = 0; kx < 5; ++kx) {
                int ix = x + kx - 2;
                if (ix < 0 || ix >= W) continue;
                s += in[((b * CIN + c) * H + iy) * W + ix] *
                     w[((o * CIN + c) * 5 + ky) * 5 + kx];
            }
        }
    out[idx] = fmaxf(s, 0.f);
}

__global__ void fc_kernel(const float* __restrict__ in, const float* __restrict__ w,
                          const float* __restrict__ bias, float* __restrict__ out,
                          int IN, int OUT, int Bn, int relu)
{
    int idx = blockIdx.x * blockDim.x + threadIdx.x;
    if (idx >= Bn * OUT) return;
    int o = idx % OUT;
    int b = idx / OUT;
    float s = bias[o];
    for (int i = 0; i < IN; ++i) s += in[b * IN + i] * w[o * IN + i];
    out[idx] = relu ? fmaxf(s, 0.f) : s;
}

// ---------------------------------------------------------------------------
extern "C" void kernel_launch(void* const* d_in, const int* in_sizes, int n_in,
                              void* d_out, int out_size, void* d_ws, size_t ws_size,
                              hipStream_t stream)
{
    (void)in_sizes; (void)n_in; (void)out_size; (void)ws_size;
    const float* x    = (const float*)d_in[0];
    const float* w1   = (const float*)d_in[1];
    const float* w2   = (const float*)d_in[2];
    const float* bn1g = (const float*)d_in[3],  *bn1b = (const float*)d_in[4];
    const float* bn1m = (const float*)d_in[5],  *bn1v = (const float*)d_in[6];
    const float* bn2g = (const float*)d_in[7],  *bn2b = (const float*)d_in[8];
    const float* bn2m = (const float*)d_in[9],  *bn2v = (const float*)d_in[10];
    const float* c3w  = (const float*)d_in[11], *c3b  = (const float*)d_in[12];
    const float* bn3g = (const float*)d_in[13], *bn3b = (const float*)d_in[14];
    const float* bn3m = (const float*)d_in[15], *bn3v = (const float*)d_in[16];
    const float* c4w  = (const float*)d_in[17], *c4b  = (const float*)d_in[18];
    const float* bn4g = (const float*)d_in[19], *bn4b = (const float*)d_in[20];
    const float* bn4m = (const float*)d_in[21], *bn4v = (const float*)d_in[22];
    const float* c5w  = (const float*)d_in[23], *c5b  = (const float*)d_in[24];
    const float* bn5g = (const float*)d_in[25], *bn5b = (const float*)d_in[26];
    const float* bn5m = (const float*)d_in[27], *bn5v = (const float*)d_in[28];
    const float* fc1w = (const float*)d_in[29], *fc1b = (const float*)d_in[30];
    const float* fc2w = (const float*)d_in[31], *fc2b = (const float*)d_in[32];
    float* out = (float*)d_out;

    char* ws = (char*)d_ws;
    size_t off = 0;
    auto alloc = [&](size_t bytes) -> char* {
        char* p = ws + off;
        off = (off + bytes + 255) & ~(size_t)255;
        return p;
    };
    float*    drive1 = (float*)   alloc(8 * 1024 * 16 * 4);
    float*    u1     = (float*)   alloc(8 * 1024 * 16 * 4);
    _Float16* gp1    = (_Float16*)alloc(1 * 41 * 512 * 2);
    float*    a1     = (float*)   alloc(8 * 16 * 1024 * 4);
    float*    xb1    = (float*)   alloc(8 * 16 * 1024 * 4);
    float*    drive2 = (float*)   alloc(8 * 1024 * 32 * 4);
    float*    u2     = (float*)   alloc(8 * 1024 * 32 * 4);
    _Float16* gp2    = (_Float16*)alloc(2 * 81 * 512 * 2);
    float*    a2     = (float*)   alloc(8 * 32 * 1024 * 4);
    float*    y2     = (float*)   alloc(8 * 32 * 256 * 4);
    float*    t3     = (float*)   alloc(8 * 64 * 256 * 4);
    float*    p3     = (float*)   alloc(8 * 64 * 64 * 4);
    float*    t4     = (float*)   alloc(8 * 128 * 64 * 4);
    float*    p4     = (float*)   alloc(8 * 128 * 16 * 4);
    float*    t5     = (float*)   alloc(8 * 256 * 16 * 4);
    float*    p5     = (float*)   alloc(8 * 256 * 4 * 4);
    float*    h1     = (float*)   alloc(8 * 500 * 4);

    hipMemsetAsync(u1, 0, 8 * 1024 * 16 * 4, stream);
    hipMemsetAsync(u2, 0, 8 * 1024 * 32 * 4, stream);

    // ---- LCA layer 1 (3 -> 16) ----
    pack_g_kernel<16, 3><<<(41 * 512 + 255) / 256, 256, 0, stream>>>(w1, gp1);
    conv_drive_kernel<3, 16><<<(8 * 1024 * 16 + 255) / 256, 256, 0, stream>>>(x, w1, drive1);
    lca_kernel<16><<<8, 256, 0, stream>>>(drive1, gp1, u1, a1);
    bn_kernel<<<(8 * 16 * 1024 + 255) / 256, 256, 0, stream>>>(a1, bn1g, bn1b, bn1m, bn1v,
                                                               xb1, 16, 1024, 8 * 16 * 1024);

    // ---- LCA layer 2 (16 -> 32) ----
    pack_g_kernel<32, 16><<<(2 * 81 * 512 + 255) / 256, 256, 0, stream>>>(w2, gp2);
    conv_drive_kernel<16, 32><<<(8 * 1024 * 32 + 255) / 256, 256, 0, stream>>>(xb1, w2, drive2);
    lca_kernel<32><<<8, 256, 0, stream>>>(drive2, gp2, u2, a2);
    pool_bn_kernel<<<(8 * 32 * 256 + 255) / 256, 256, 0, stream>>>(a2, bn2g, bn2b, bn2m, bn2v,
                                                                   y2, 32, 32, 32, 8 * 32 * 256);

    // ---- conv blocks ----
    conv_bias_relu_kernel<<<(8 * 64 * 256 + 255) / 256, 256, 0, stream>>>(y2, c3w, c3b, t3, 32, 64, 16, 16, 8 * 64 * 256);
    pool_bn_kernel<<<(8 * 64 * 64 + 255) / 256, 256, 0, stream>>>(t3, bn3g, bn3b, bn3m, bn3v, p3, 64, 16, 16, 8 * 64 * 64);
    conv_bias_relu_kernel<<<(8 * 128 * 64 + 255) / 256, 256, 0, stream>>>(p3, c4w, c4b, t4, 64, 128, 8, 8, 8 * 128 * 64);
    pool_bn_kernel<<<(8 * 128 * 16 + 255) / 256, 256, 0, stream>>>(t4, bn4g, bn4b, bn4m, bn4v, p4, 128, 8, 8, 8 * 128 * 16);
    conv_bias_relu_kernel<<<(8 * 256 * 16 + 255) / 256, 256, 0, stream>>>(p4, c5w, c5b, t5, 128, 256, 4, 4, 8 * 256 * 16);
    pool_bn_kernel<<<(8 * 256 * 4 + 255) / 256, 256, 0, stream>>>(t5, bn5g, bn5b, bn5m, bn5v, p5, 256, 4, 4, 8 * 256 * 4);

    // ---- classifier ----
    fc_kernel<<<(8 * 500 + 255) / 256, 256, 0, stream>>>(p5, fc1w, fc1b, h1, 1024, 500, 8, 1);
    fc_kernel<<<1, 64, 0, stream>>>(h1, fc2w, fc2b, out, 500, 5, 8, 0);
}
